// FlexSummaryAttention_74629351735342
// MI455X (gfx1250) — matrix operand order repaired and checked by reading, not compiled, not run on hardware
//
#include <hip/hip_runtime.h>
#include <hip/hip_bf16.h>
#include <math.h>
#include <stdint.h>

typedef unsigned short u16;
typedef __attribute__((ext_vector_type(16))) __bf16       bf16x16;
typedef __attribute__((ext_vector_type(8)))  float        f32x8;
typedef __attribute__((ext_vector_type(4)))  unsigned int u32x4;
typedef __attribute__((ext_vector_type(8)))  int          i32x8;
typedef __attribute__((ext_vector_type(4)))  int          i32x4;

constexpr int EMB   = 1024;
constexpr int NH    = 16;
constexpr int HD    = 64;
constexpr int NB    = 2;
constexpr int SL    = 64;
constexpr int RL    = 2048;
constexpr int BARSZ = 32;
constexpr int NBARS = RL / BARSZ;   // 64
constexpr int LK1   = SL + RL;      // 2112

#if __has_builtin(__builtin_amdgcn_tensor_load_to_lds)
#define HAVE_TDM 1
#else
#define HAVE_TDM 0
#endif

// ---------------- helpers ----------------

__device__ __forceinline__ u16 f2bf(float f) {
  unsigned u = __float_as_uint(f);
  u += 0x7fffu + ((u >> 16) & 1u);            // round-to-nearest-even
  return (u16)(u >> 16);
}

__device__ __forceinline__ f32x8 zero8() {
  f32x8 z;
#pragma unroll
  for (int i = 0; i < 8; i++) z[i] = 0.f;
  return z;
}

// assemble 16 bf16 lanes-worth from two 8-element contiguous runs
__device__ __forceinline__ bf16x16 ld16(const u16* p0, const u16* p1) {
  union { u32x4 q[2]; bf16x16 v; } t;
  t.q[0] = *reinterpret_cast<const u32x4*>(p0);
  t.q[1] = *reinterpret_cast<const u32x4*>(p1);
  return t.v;
}

// A operand (16x32, row-major source, ld elements/row):
// lane m=lane&15, h=lane>>4; VGPR0-7: K = 16h .. 16h+15 (contiguous)
__device__ __forceinline__ bf16x16 load_a(const u16* base, int ld, int m0, int k0) {
  int lane = threadIdx.x & 31;
  int m = lane & 15, h = lane >> 4;
  const u16* p = base + (size_t)(m0 + m) * ld + k0 + 16 * h;
  return ld16(p, p + 8);
}

// B operand (32x16).  Source laid out (N,K) row-major (ld elements/row):
// lane n=lane&15, h=lane>>4 ; VGPR0-7: K = 16h .. 16h+15 (contiguous)
__device__ __forceinline__ bf16x16 load_b(const u16* base, int ld, int n0, int k0) {
  int lane = threadIdx.x & 31;
  int n = lane & 15, h = lane >> 4;
  const u16* p = base + (size_t)(n0 + n) * ld + k0 + 16 * h;
  return ld16(p, p + 8);
}

__device__ __forceinline__ f32x8 wmma_bf16(bf16x16 a, bf16x16 b, f32x8 c) {
  return __builtin_amdgcn_wmma_f32_16x16x32_bf16(false, a, false, b, (short)0, c, false, false);
}

// reductions across a 16-lane half-wave (one C-matrix row)
__device__ __forceinline__ float redmax16(float x) {
#pragma unroll
  for (int m = 1; m < 16; m <<= 1) x = fmaxf(x, __shfl_xor(x, m, 32));
  return x;
}
__device__ __forceinline__ float redsum16(float x) {
#pragma unroll
  for (int m = 1; m < 16; m <<= 1) x += __shfl_xor(x, m, 32);
  return x;
}

#if HAVE_TDM
// ---- Tensor Data Mover: 2-D bf16 tile (rows x kelems) global -> LDS ----
// D# per CDNA5 ISA (2-D, groups 2/3 zero).
__device__ __forceinline__ void tdm_load_2d(const u16* gsrc, unsigned lds_off,
                                            int kelems, int rows,
                                            int row_stride_elems,
                                            int tensor_rows) {
  unsigned long long ga = (unsigned long long)(uintptr_t)gsrc;
  u32x4 g0;
  g0[0] = 1u;                                             // count=1 (valid), user mode
  g0[1] = lds_off;                                        // lds_addr (bytes)
  g0[2] = (unsigned)(ga & 0xffffffffu);                   // global_addr[31:0]
  g0[3] = (unsigned)((ga >> 32) & 0x1ffffffu) | (2u << 30); // global_addr[56:32] | type=2
  unsigned td0 = (unsigned)row_stride_elems;              // tensor dim0 (elements)
  unsigned td1 = (unsigned)tensor_rows;                   // tensor dim1 (rows)
  i32x8 g1;
  g1[0] = (int)(1u << 16);                                // data_size=1 -> 2 bytes/elem
  g1[1] = (int)((td0 & 0xffffu) << 16);                   // tensor_dim0[15:0] @ bits 79:48
  g1[2] = (int)((td0 >> 16) | ((td1 & 0xffffu) << 16));   // dim0 hi | tensor_dim1 lo
  g1[3] = (int)((td1 >> 16) | ((unsigned)kelems << 16));  // dim1 hi | tile_dim0
  g1[4] = (int)((unsigned)rows & 0xffffu);                // tile_dim1 (tile_dim2 = 0)
  g1[5] = (int)(unsigned)row_stride_elems;                // tensor_dim0_stride[31:0]
  g1[6] = 0;                                              // stride hi | dim1_stride lo
  g1[7] = 0;
  i32x4 z4 = {0, 0, 0, 0};
#if __clang_major__ >= 23
  i32x8 z8 = {0, 0, 0, 0, 0, 0, 0, 0};
  __builtin_amdgcn_tensor_load_to_lds(g0, g1, z4, z4, z8, 0);
#else
  __builtin_amdgcn_tensor_load_to_lds(g0, g1, z4, z4, 0);
#endif
}
#endif

__device__ __forceinline__ unsigned lds_off_of(const void* p) {
  return (unsigned)(uintptr_t)p;     // LDS aperture occupies the high 32 bits
}

// ---------------- elementwise / layout kernels ----------------

__global__ void k_f32_to_bf16(const float* __restrict__ x, u16* __restrict__ y, int n) {
  int i = blockIdx.x * blockDim.x + threadIdx.x;
  if (i < n) y[i] = f2bf(x[i]);
}

// W (K=1024, N=1024) fp32  ->  Wt (N, K) bf16
__global__ void k_wtrans_bf16(const float* __restrict__ W, u16* __restrict__ Wt) {
  __shared__ float tile[32][33];
  int bn = blockIdx.x * 32, bk = blockIdx.y * 32;
  int tx = threadIdx.x, ty = threadIdx.y;           // (32, 8)
#pragma unroll
  for (int i = 0; i < 32; i += 8)
    tile[ty + i][tx] = W[(size_t)(bk + ty + i) * EMB + bn + tx];
  __syncthreads();
#pragma unroll
  for (int i = 0; i < 32; i += 8)
    Wt[(size_t)(bn + ty + i) * EMB + bk + tx] = f2bf(tile[tx][ty + i]);
}

// q,k: (NB, L, EMB) fp32 -> RoPE(pos0+l) -> (NB, NH, L, HD) bf16 row-major
__global__ void k_rope_heads(const float* __restrict__ q, const float* __restrict__ k,
                             int L, int pos0, u16* __restrict__ qh, u16* __restrict__ kh) {
  int i = blockIdx.x * blockDim.x + threadIdx.x;
  int total = NB * L * EMB;
  if (i >= total) return;
  int c = i % EMB; int bl = i / EMB; int l = bl % L; int b = bl / L;
  int h = c >> 6, d = c & 63;
  int fi = d & 31;
  float inv = __expf(-(float)(2 * fi) * (1.0f / 64.0f) * 9.210340372f); // ln(10000)
  float ang = (float)(pos0 + l) * inv;
  float sn, cs; __sincosf(ang, &sn, &cs);
  float qv = q[i], kv = k[i];
  int pidx = (d < 32) ? (i + 32) : (i - 32);
  float qr = (d < 32) ? -q[pidx] : q[pidx];
  float kr = (d < 32) ? -k[pidx] : k[pidx];
  size_t o = ((size_t)(b * NH + h) * L + l) * HD + d;
  qh[o] = f2bf(qv * cs + qr * sn);
  kh[o] = f2bf(kv * cs + kr * sn);
}

// v: (NB, L, EMB) fp32 -> (NB, NH, HD, L) bf16   (B-operand layout for P·V)
__global__ void k_v_heads_t(const float* __restrict__ v, int L, u16* __restrict__ vt) {
  int i = blockIdx.x * blockDim.x + threadIdx.x;
  int total = NB * L * EMB;
  if (i >= total) return;
  int c = i % EMB; int bl = i / EMB; int l = bl % L; int b = bl / L;
  int h = c >> 6, d = c & 63;
  vt[((size_t)(b * NH + h) * HD + d) * L + l] = f2bf(v[i]);
}

// k2,v2: (NB, SL, EMB) fp32 -> k2h (NB,NH,SL,HD) bf16 rm ; v2t (NB,NH,HD,SL) bf16
__global__ void k_split_k2v2(const float* __restrict__ k2, const float* __restrict__ v2,
                             u16* __restrict__ k2h, u16* __restrict__ v2t) {
  int i = blockIdx.x * blockDim.x + threadIdx.x;
  int total = NB * SL * EMB;
  if (i >= total) return;
  int c = i % EMB; int bl = i / EMB; int l = bl % SL; int b = bl / SL;
  int h = c >> 6, d = c & 63;
  k2h[((size_t)(b * NH + h) * SL + l) * HD + d] = f2bf(k2[i]);
  v2t[((size_t)(b * NH + h) * HD + d) * SL + l] = f2bf(v2[i]);
}

// ---------------- GEMM: C(M,1024) = A(M,1024)bf16 @ Wt(N,K)bf16 + bias ----------------
// block = 256 threads (8 waves). Block tile 64(M) x 128(N); wave tile 32x32.
// A/B tiles (64x64 / 128x64 bf16) staged in LDS by the Tensor Data Mover,
// double buffered, synchronized with s_wait_tensorcnt + workgroup barriers.
__global__ void k_gemm(const u16* __restrict__ A, const u16* __restrict__ Wt,
                       const float* __restrict__ bias, float* __restrict__ C, int M) {
  __shared__ __align__(16) u16 As[2][64 * 64];
  __shared__ __align__(16) u16 Bs[2][128 * 64];
  int wave = threadIdx.x >> 5;
  int lane = threadIdx.x & 31;
  int mw = wave & 1, nw = wave >> 1;            // waves: 2 (M) x 4 (N)
  int mBlk = blockIdx.x * 64, nBlk = blockIdx.y * 128;
  int m0 = mw * 32, n0 = nw * 32;               // offsets within LDS tiles
  f32x8 acc00 = zero8(), acc01 = zero8(), acc10 = zero8(), acc11 = zero8();
  constexpr int KSTEPS = EMB / 64;              // 16

#if HAVE_TDM
  if (wave == 0) {
    tdm_load_2d(A  + (size_t)mBlk * EMB,      lds_off_of(As[0]), 64, 64,  EMB, M);
    tdm_load_2d(Wt + (size_t)nBlk * EMB,      lds_off_of(Bs[0]), 64, 128, EMB, EMB);
    tdm_load_2d(A  + (size_t)mBlk * EMB + 64, lds_off_of(As[1]), 64, 64,  EMB, M);
    tdm_load_2d(Wt + (size_t)nBlk * EMB + 64, lds_off_of(Bs[1]), 64, 128, EMB, EMB);
  }
#endif
  for (int ks = 0; ks < KSTEPS; ks++) {
    int cur = ks & 1;
#if HAVE_TDM
    if (wave == 0) {
      if (ks + 1 < KSTEPS) __builtin_amdgcn_s_wait_tensorcnt(2);  // next prefetch may fly
      else                 __builtin_amdgcn_s_wait_tensorcnt(0);
    }
    __syncthreads();
#else
    { // cooperative staging fallback
      int k0 = ks * 64;
      int t = threadIdx.x;
      for (int j = t; j < 512; j += 256) {      // A: 64 rows x 8 uint4
        int row = j >> 3, c16 = (j & 7) * 8;
        *reinterpret_cast<u32x4*>(&As[cur][row * 64 + c16]) =
            *reinterpret_cast<const u32x4*>(A + (size_t)(mBlk + row) * EMB + k0 + c16);
      }
      for (int j = t; j < 1024; j += 256) {     // B: 128 rows x 8 uint4
        int row = j >> 3, c16 = (j & 7) * 8;
        *reinterpret_cast<u32x4*>(&Bs[cur][row * 64 + c16]) =
            *reinterpret_cast<const u32x4*>(Wt + (size_t)(nBlk + row) * EMB + k0 + c16);
      }
      __syncthreads();
    }
#endif
    const u16* At = As[cur];
    const u16* Bt = Bs[cur];
#pragma unroll
    for (int kk = 0; kk < 64; kk += 32) {
      bf16x16 a0 = load_a(At, 64, m0, kk);
      bf16x16 a1 = load_a(At, 64, m0 + 16, kk);
      bf16x16 b0 = load_b(Bt, 64, n0, kk);
      bf16x16 b1 = load_b(Bt, 64, n0 + 16, kk);
      acc00 = wmma_bf16(a0, b0, acc00);
      acc01 = wmma_bf16(a0, b1, acc01);
      acc10 = wmma_bf16(a1, b0, acc10);
      acc11 = wmma_bf16(a1, b1, acc11);
    }
    __syncthreads();
#if HAVE_TDM
    if (wave == 0 && ks + 2 < KSTEPS) {
      int k0 = (ks + 2) * 64;
      tdm_load_2d(A  + (size_t)mBlk * EMB + k0, lds_off_of(As[cur]), 64, 64,  EMB, M);
      tdm_load_2d(Wt + (size_t)nBlk * EMB + k0, lds_off_of(Bs[cur]), 64, 128, EMB, EMB);
    }
#endif
  }
  int n = lane & 15, h = lane >> 4;
  int gm0 = mBlk + m0, gn0 = nBlk + n0;
  float b0v = bias[gn0 + n], b1v = bias[gn0 + 16 + n];
#pragma unroll
  for (int v = 0; v < 8; v++) {
    int m = v + 8 * h;
    C[(size_t)(gm0 + m) * EMB + gn0 + n]           = acc00[v] + b0v;
    C[(size_t)(gm0 + m) * EMB + gn0 + 16 + n]      = acc01[v] + b1v;
    C[(size_t)(gm0 + 16 + m) * EMB + gn0 + n]      = acc10[v] + b0v;
    C[(size_t)(gm0 + 16 + m) * EMB + gn0 + 16 + n] = acc11[v] + b1v;
  }
}

// ---------------- Stage 1: summaries attend to [sum ; reg] ----------------

// one wave per 16x16 score tile. grid (132 ktiles, 4 qtiles, 32 bh), block 32
__global__ void k_s1_scores(const u16* __restrict__ sqh, const u16* __restrict__ skh,
                            const u16* __restrict__ rkh, float* __restrict__ sc) {
  int kt = blockIdx.x, qt = blockIdx.y, bh = blockIdx.z;
  const u16* qb = sqh + (size_t)bh * SL * HD + qt * 16 * HD;
  const u16* kb = (kt < 4) ? skh + (size_t)bh * SL * HD + kt * 16 * HD
                           : rkh + (size_t)bh * RL * HD + (size_t)(kt - 4) * 16 * HD;
  f32x8 c = zero8();
  c = wmma_bf16(load_a(qb, HD, 0, 0),  load_b(kb, HD, 0, 0),  c);
  c = wmma_bf16(load_a(qb, HD, 0, 32), load_b(kb, HD, 0, 32), c);
  int lane = threadIdx.x & 31;
  int n = lane & 15, h = lane >> 4;
#pragma unroll
  for (int v = 0; v < 8; v++) {
    int m = v + 8 * h;
    int qrow = qt * 16 + m;
    bool valid;
    if (kt < 4) valid = (kt * 16 + n) <= qrow;                    // causal over summaries
    else { int r = (kt - 4) * 16 + n; valid = (r >> 5) == qrow; } // bar_of(r) == q
    sc[((size_t)bh * SL + qrow) * LK1 + kt * 16 + n] = valid ? c[v] * 0.125f : -3.0e38f;
  }
}

// row softmax: grid 2048 rows, block 256; fp32 in, bf16 probs out
__global__ void k_s1_softmax(const float* __restrict__ sc, u16* __restrict__ p) {
  __shared__ float red[8];
  int row = blockIdx.x;
  const float* x = sc + (size_t)row * LK1;
  u16* y = p + (size_t)row * LK1;
  int t = threadIdx.x;
  float mx = -3.0e38f;
  for (int i = t; i < LK1; i += 256) mx = fmaxf(mx, x[i]);
#pragma unroll
  for (int m = 1; m < 32; m <<= 1) mx = fmaxf(mx, __shfl_xor(mx, m, 32));
  if ((t & 31) == 0) red[t >> 5] = mx;
  __syncthreads();
  mx = red[0];
#pragma unroll
  for (int i = 1; i < 8; i++) mx = fmaxf(mx, red[i]);
  float sum = 0.f;
  for (int i = t; i < LK1; i += 256) sum += __expf(x[i] - mx);
#pragma unroll
  for (int m = 1; m < 32; m <<= 1) sum += __shfl_xor(sum, m, 32);
  __syncthreads();
  if ((t & 31) == 0) red[t >> 5] = sum;
  __syncthreads();
  sum = 0.f;
#pragma unroll
  for (int i = 0; i < 8; i++) sum += red[i];
  float inv = 1.0f / sum;
  for (int i = t; i < LK1; i += 256) y[i] = f2bf(__expf(x[i] - mx) * inv);
}

// P(64x2112) @ V(2112x64): one wave per 16x16 out tile; writes merged-head sum_attn
// grid (4 dtiles, 4 qtiles, 32 bh), block 32
__global__ void k_s1_pv(const u16* __restrict__ p, const u16* __restrict__ svt,
                        const u16* __restrict__ rvt, float* __restrict__ sum_attn) {
  int dt = blockIdx.x, qt = blockIdx.y, bh = blockIdx.z;
  int b = bh >> 4, h = bh & 15;
  const u16* pb = p + (size_t)bh * SL * LK1 + (size_t)qt * 16 * LK1;
  f32x8 acc = zero8();
  for (int s = 0; s < LK1 / 32; s++) {              // 66 K-steps
    bf16x16 a = load_a(pb, LK1, 0, s * 32);
    const u16* vb; int ld, k0;
    if (s < 2) { vb = svt + ((size_t)bh * HD + dt * 16) * SL; ld = SL; k0 = s * 32; }
    else       { vb = rvt + ((size_t)bh * HD + dt * 16) * RL; ld = RL; k0 = s * 32 - SL; }
    acc = wmma_bf16(a, load_b(vb, ld, 0, k0), acc);
  }
  int lane = threadIdx.x & 31;
  int n = lane & 15, hh = lane >> 4;
#pragma unroll
  for (int v = 0; v < 8; v++) {
    int m = v + 8 * hh;
    sum_attn[((size_t)b * SL + qt * 16 + m) * EMB + h * HD + dt * 16 + n] = acc[v];
  }
}

// ---------------- Stage 2: fused block-sparse attention ----------------
// grid (64 bars, 16 heads, 2 batch), block 64 (2 waves; wave = 16 query rows)
__global__ void k_s2_attn(const u16* __restrict__ rqh, const u16* __restrict__ k2h,
                          const u16* __restrict__ rkh, const u16* __restrict__ v2t,
                          const u16* __restrict__ rvt, float* __restrict__ reg_attn) {
  __shared__ __align__(16) u16 P[2][16][96];
  int bar = blockIdx.x, h = blockIdx.y, b = blockIdx.z;
  int w = threadIdx.x >> 5, lane = threadIdx.x & 31;
  int bh = b * NH + h;
  int nst  = (bar >> 4) + 1;          // summary tiles needed (ceil((bar+1)/16))
  int nst2 = (nst + 1) & ~1;          // pad to even (2 or 4) so K-steps stay in one source
  int nkt  = nst2 + 2;                // + 2 regular tiles (own bar) => 4 or 6
  int q0 = bar * BARSZ + w * 16;
  const u16* qb = rqh + ((size_t)bh * RL + q0) * HD;
  bf16x16 a0 = load_a(qb, HD, 0, 0);
  bf16x16 a1 = load_a(qb, HD, 0, 32);

  f32x8 s[6];
#pragma unroll
  for (int t = 0; t < 6; t++) {
    if (t < nkt) {
      const u16* kb = (t < nst2)
          ? k2h + ((size_t)bh * SL + t * 16) * HD
          : rkh + ((size_t)bh * RL + bar * BARSZ + (t - nst2) * 16) * HD;
      f32x8 c = zero8();
      c = wmma_bf16(a0, load_b(kb, HD, 0, 0),  c);
      c = wmma_bf16(a1, load_b(kb, HD, 0, 32), c);
      s[t] = c;
    }
  }

  int n = lane & 15, hh = lane >> 4;
#pragma unroll
  for (int v = 0; v < 8; v++) {
    int m = v + 8 * hh;
    int qloc = w * 16 + m;                 // query index within the bar (0..31)
    float mx = -3.0e38f;
#pragma unroll
    for (int t = 0; t < 6; t++) {
      if (t < nkt) {
        bool valid = (t < nst2) ? ((t * 16 + n) <= bar)
                                : (((t - nst2) * 16 + n) <= qloc);
        float x = valid ? s[t][v] * 0.125f : -3.0e38f;
        s[t][v] = x;
        mx = fmaxf(mx, x);
      }
    }
    mx = redmax16(mx);
    float sum = 0.f;
#pragma unroll
    for (int t = 0; t < 6; t++) {
      if (t < nkt) { float e = __expf(s[t][v] - mx); s[t][v] = e; sum += e; }
    }
    sum = redsum16(sum);
    float inv = 1.0f / sum;
#pragma unroll
    for (int t = 0; t < 6; t++) {
      if (t < nkt) P[w][m][t * 16 + n] = f2bf(s[t][v] * inv);
    }
  }
  __syncthreads();

  f32x8 acc[4] = {zero8(), zero8(), zero8(), zero8()};
  int ksteps   = nkt >> 1;     // 2 or 3
  int sumsteps = nst2 >> 1;    // 1 or 2
#pragma unroll
  for (int ss = 0; ss < 3; ss++) {
    if (ss < ksteps) {
      bf16x16 ap = load_a(&P[w][0][0], 96, 0, ss * 32);
#pragma unroll
      for (int dt = 0; dt < 4; dt++) {
        const u16* vb; int ld, k0;
        if (ss < sumsteps) { vb = v2t + ((size_t)bh * HD + dt * 16) * SL; ld = SL; k0 = ss * 32; }
        else { vb = rvt + ((size_t)bh * HD + dt * 16) * RL; ld = RL; k0 = bar * BARSZ + (ss - sumsteps) * 32; }
        acc[dt] = wmma_bf16(ap, load_b(vb, ld, 0, k0), acc[dt]);
      }
    }
  }
#pragma unroll
  for (int dt = 0; dt < 4; dt++)
#pragma unroll
    for (int v = 0; v < 8; v++) {
      int m = v + 8 * hh;
      reg_attn[((size_t)b * RL + q0 + m) * EMB + h * HD + dt * 16 + n] = acc[dt][v];
    }
}

// ---------------- host-side orchestration ----------------

extern "C" void kernel_launch(void* const* d_in, const int* in_sizes, int n_in,
                              void* d_out, int out_size, void* d_ws, size_t ws_size,
                              hipStream_t stream) {
  (void)in_sizes; (void)n_in; (void)out_size; (void)ws_size;
  const float* sum_x = (const float*)d_in[0];
  const float* reg_x = (const float*)d_in[1];
  // masks d_in[2], d_in[3] are analytic functions of indices -> recomputed on device
  const float* W[10]; const float* Bv[10];
  for (int i = 0; i < 10; i++) { W[i] = (const float*)d_in[4 + 2 * i]; Bv[i] = (const float*)d_in[5 + 2 * i]; }
  // order: 0 wq_s 1 wk_s 2 wv_s 3 wo_s 4 wq_r 5 wk_r 6 wv_r 7 wo_r 8 wk2 9 wv2
  float* out = (float*)d_out;

  char* ws = (char*)d_ws;
  size_t off = 0;
  auto alloc = [&](size_t bytes) -> void* {
    off = (off + 255) & ~(size_t)255;
    void* p = ws + off;
    off += bytes;
    return p;
  };

  const size_t nSum = (size_t)NB * SL * EMB;   // 131072
  const size_t nReg = (size_t)NB * RL * EMB;   // 4194304

  u16* Wt[10];
  for (int i = 0; i < 10; i++) Wt[i] = (u16*)alloc((size_t)EMB * EMB * sizeof(u16));
  u16* xs_bf = (u16*)alloc(nSum * sizeof(u16));
  u16* xr_bf = (u16*)alloc(nReg * sizeof(u16));
  u16* sqh   = (u16*)alloc(nSum * sizeof(u16));
  u16* skh   = (u16*)alloc(nSum * sizeof(u16));
  u16* svt   = (u16*)alloc(nSum * sizeof(u16));
  u16* rqh   = (u16*)alloc(nReg * sizeof(u16));
  u16* rkh   = (u16*)alloc(nReg * sizeof(u16));
  u16* rvt   = (u16*)alloc(nReg * sizeof(u16));
  u16* k2h   = (u16*)alloc(nSum * sizeof(u16));
  u16* v2t   = (u16*)alloc(nSum * sizeof(u16));
  u16* p1    = (u16*)alloc((size_t)NB * NH * SL * LK1 * sizeof(u16));
  u16* sa_bf = (u16*)alloc(nSum * sizeof(u16));
  u16* ra_bf = (u16*)alloc(nReg * sizeof(u16));
  float* sum_attn = (float*)alloc(nSum * sizeof(float));
  float* tmpk2    = (float*)alloc(nSum * sizeof(float));
  float* tmpv2    = (float*)alloc(nSum * sizeof(float));
  // 32MB phased region: {tmpA,tmpB} -> scores1 -> reg_attn
  float* big  = (float*)alloc(2 * nReg * sizeof(float) > (size_t)NB * NH * SL * LK1 * sizeof(float)
                              ? 2 * nReg * sizeof(float)
                              : (size_t)NB * NH * SL * LK1 * sizeof(float));
  float* tmpA = big;
  float* tmpB = big + nReg;
  float* scores1  = big;
  float* reg_attn = big;

  dim3 b256(256);
  auto grid1 = [](size_t n) { return dim3((unsigned)((n + 255) / 256)); };
  dim3 wtg(EMB / 32, EMB / 32), wtb(32, 8);
  dim3 gemm_sum(NB * SL / 64, EMB / 128);   // M=128 : (2, 8)
  dim3 gemm_reg(NB * RL / 64, EMB / 128);   // M=4096: (64, 8)

  // phase 0: precision/layout conversion
  hipLaunchKernelGGL(k_f32_to_bf16, grid1(nSum), b256, 0, stream, sum_x, xs_bf, (int)nSum);
  hipLaunchKernelGGL(k_f32_to_bf16, grid1(nReg), b256, 0, stream, reg_x, xr_bf, (int)nReg);
  for (int i = 0; i < 10; i++)
    hipLaunchKernelGGL(k_wtrans_bf16, wtg, wtb, 0, stream, W[i], Wt[i]);

  // phase 1: summary projections + RoPE
  hipLaunchKernelGGL(k_gemm, gemm_sum, b256, 0, stream, xs_bf, Wt[0], Bv[0], tmpA, NB * SL);
  hipLaunchKernelGGL(k_gemm, gemm_sum, b256, 0, stream, xs_bf, Wt[1], Bv[1], tmpB, NB * SL);
  hipLaunchKernelGGL(k_rope_heads, grid1(nSum), b256, 0, stream, tmpA, tmpB, SL, 0, sqh, skh);
  hipLaunchKernelGGL(k_gemm, gemm_sum, b256, 0, stream, xs_bf, Wt[2], Bv[2], tmpA, NB * SL);
  hipLaunchKernelGGL(k_v_heads_t, grid1(nSum), b256, 0, stream, tmpA, SL, svt);

  // phase 2: regular projections + RoPE (positions S..S+R)
  hipLaunchKernelGGL(k_gemm, gemm_reg, b256, 0, stream, xr_bf, Wt[4], Bv[4], tmpA, NB * RL);
  hipLaunchKernelGGL(k_gemm, gemm_reg, b256, 0, stream, xr_bf, Wt[5], Bv[5], tmpB, NB * RL);
  hipLaunchKernelGGL(k_rope_heads, grid1(nReg), b256, 0, stream, tmpA, tmpB, RL, SL, rqh, rkh);
  hipLaunchKernelGGL(k_gemm, gemm_reg, b256, 0, stream, xr_bf, Wt[6], Bv[6], tmpA, NB * RL);
  hipLaunchKernelGGL(k_v_heads_t, grid1(nReg), b256, 0, stream, tmpA, RL, rvt);

  // phase 3: stage-1 attention (tmpA/tmpB dead -> reuse `big` as score matrix)
  hipLaunchKernelGGL(k_s1_scores, dim3(LK1 / 16, SL / 16, NB * NH), dim3(32), 0, stream,
                     sqh, skh, rkh, scores1);
  hipLaunchKernelGGL(k_s1_softmax, dim3(NB * NH * SL), b256, 0, stream, scores1, p1);
  hipLaunchKernelGGL(k_s1_pv, dim3(HD / 16, SL / 16, NB * NH), dim3(32), 0, stream,
                     p1, svt, rvt, sum_attn);

  // phase 4: k2/v2 re-projection + summary output projection
  hipLaunchKernelGGL(k_f32_to_bf16, grid1(nSum), b256, 0, stream, sum_attn, sa_bf, (int)nSum);
  hipLaunchKernelGGL(k_gemm, gemm_sum, b256, 0, stream, sa_bf, Wt[8], Bv[8], tmpk2, NB * SL);
  hipLaunchKernelGGL(k_gemm, gemm_sum, b256, 0, stream, sa_bf, Wt[9], Bv[9], tmpv2, NB * SL);
  hipLaunchKernelGGL(k_split_k2v2, grid1(nSum), b256, 0, stream, tmpk2, tmpv2, k2h, v2t);
  hipLaunchKernelGGL(k_gemm, gemm_sum, b256, 0, stream, sa_bf, Wt[3], Bv[3], out, NB * SL);

  // phase 5: stage-2 block-sparse attention + regular output projection
  hipLaunchKernelGGL(k_s2_attn, dim3(NBARS, NH, NB), dim3(64), 0, stream,
                     rqh, k2h, rkh, v2t, rvt, reg_attn);
  hipLaunchKernelGGL(k_f32_to_bf16, grid1(nReg), b256, 0, stream, reg_attn, ra_bf, (int)nReg);
  hipLaunchKernelGGL(k_gemm, gemm_reg, b256, 0, stream, ra_bf, Wt[7], Bv[7],
                     out + nSum, NB * RL);
}
